// PyramidGraphSage_89326729822493
// MI455X (gfx1250) — matrix operand order repaired and checked by reading, not compiled, not run on hardware
//
#include <hip/hip_runtime.h>
#include <math.h>

// ---------------------------------------------------------------------------
// PyramidGraphSage on gfx1250 (MI455X): all GEMMs via v_wmma_f32_16x16x32_bf16
// B=2, N=2048, R=128, 3R=384. All GEMM dims are multiples of tile sizes.
// Fragments are fed by aligned ds_load_b128 (LDS B tile stored transposed).
// ---------------------------------------------------------------------------

#define BM 128
#define BN 128
#define BK 32
#define LDP (BK + 8)        // padded row: 40 ushorts = 80 B (16B-aligned rows)
#define GTHREADS 256

typedef __attribute__((ext_vector_type(16))) __bf16 v16bf;
typedef __attribute__((ext_vector_type(8)))  float  v8f;

struct Frag32 { uint4 lo, hi; };   // 32 bytes == 16 bf16

__device__ __forceinline__ unsigned short f2bf(float f) {
  unsigned u = __float_as_uint(f);
  return (unsigned short)((u + 0x7FFFu + ((u >> 16) & 1u)) >> 16);
}
__device__ __forceinline__ unsigned pack2(float a, float b) {
  return (unsigned)f2bf(a) | ((unsigned)f2bf(b) << 16);
}

struct GemmParams {
  const float* a[3]; long lda[3]; long abs_[3]; int klen[3]; int arelu[3];
  int nseg;                 // number of A segments concatenated along K
  int transA;               // A[m,k] = src[k*lda + m] (only nseg==1)
  const float* b; long ldb; long bbs; int transB;   // transB: B[k,n]=src[n*ldb+k]
  const float* bias;
  float* c; long ldc; long cbs;
  const float* aux; long ldaux; long auxbs;         // adj mask (mode 3)
  int M, N, K;
  int mode;                 // 0:none 1:relu 2:relu if col>=relu_col 3:exp*aux
  int relu_col;
};

__global__ __launch_bounds__(GTHREADS)
void wmma_gemm(GemmParams p) {
  __shared__ __align__(16) unsigned short As[BM][LDP];    // A[m][k]
  __shared__ __align__(16) unsigned short BsT[BN][LDP];   // B^T: [n][k]

  const int tid   = threadIdx.x;
  const int lane  = tid & 31;
  const int half  = lane >> 4;        // 0 | 1
  const int lrow  = lane & 15;
  const int wave  = tid >> 5;
  const int waveM = wave >> 2;        // 0..1  -> 64-row slabs
  const int waveN = wave & 3;         // 0..3  -> 32-col slabs
  const int batch = blockIdx.z;
  const long mBase = (long)blockIdx.y * BM;
  const long nBase = (long)blockIdx.x * BN;

  v8f acc[4][2];
  #pragma unroll
  for (int i = 0; i < 4; i++)
    #pragma unroll
    for (int j = 0; j < 2; j++)
      #pragma unroll
      for (int e = 0; e < 8; e++) acc[i][j][e] = 0.0f;

  for (int k0 = 0; k0 < p.K; k0 += BK) {
    // ---- resolve A segment for this K-tile (tiles never straddle segments)
    int seg = 0, kloc = k0;
    if (p.nseg > 1) {
      while (seg < p.nseg - 1 && kloc >= p.klen[seg]) { kloc -= p.klen[seg]; seg++; }
    }
    const float* aseg = p.a[seg] + (long)batch * p.abs_[seg];
    const long   lda  = p.lda[seg];
    const int    arel = p.arelu[seg];

    // ---- stage A tile (f32 -> bf16, optional relu-on-load)
    if (!p.transA) {
      // K-contiguous in memory: float4 loads, uint2 packed LDS stores
      for (int idx = tid; idx < BM * BK / 4; idx += GTHREADS) {
        int m = idx >> 3;                 // 8 float4 per row
        int k = (idx & 7) * 4;
        float4 v = *(const float4*)(aseg + (mBase + m) * lda + (kloc + k));
        if (arel) {
          v.x = fmaxf(v.x, 0.f); v.y = fmaxf(v.y, 0.f);
          v.z = fmaxf(v.z, 0.f); v.w = fmaxf(v.w, 0.f);
        }
        uint2 pk; pk.x = pack2(v.x, v.y); pk.y = pack2(v.z, v.w);
        *(uint2*)&As[m][k] = pk;
      }
    } else {
      // M-contiguous in memory (dp^T): float4 loads, 4 scalar LDS stores
      for (int idx = tid; idx < BM * BK / 4; idx += GTHREADS) {
        int k = idx >> 5;                 // 32 float4 per k-line
        int m = (idx & 31) * 4;
        float4 v = *(const float4*)(aseg + (long)(kloc + k) * lda + (mBase + m));
        As[m + 0][k] = f2bf(v.x);
        As[m + 1][k] = f2bf(v.y);
        As[m + 2][k] = f2bf(v.z);
        As[m + 3][k] = f2bf(v.w);
      }
    }
    // ---- stage B tile into transposed LDS layout BsT[n][k]
    const float* bptr = p.b + (long)batch * p.bbs;
    if (!p.transB) {
      // weights [K,N]: N-contiguous: float4 loads, scalar transposed stores
      for (int idx = tid; idx < BK * BN / 4; idx += GTHREADS) {
        int k = idx >> 5;                 // 32 float4 per k-line
        int n = (idx & 31) * 4;
        float4 v = *(const float4*)(bptr + (long)(k0 + k) * p.ldb + (nBase + n));
        BsT[n + 0][k] = f2bf(v.x);
        BsT[n + 1][k] = f2bf(v.y);
        BsT[n + 2][k] = f2bf(v.z);
        BsT[n + 3][k] = f2bf(v.w);
      }
    } else {
      // na [N,K]: K-contiguous: float4 loads, uint2 packed LDS stores
      for (int idx = tid; idx < BK * BN / 4; idx += GTHREADS) {
        int n = idx >> 3;                 // 8 float4 per n-row
        int k = (idx & 7) * 4;
        float4 v = *(const float4*)(bptr + (long)(nBase + n) * p.ldb + (k0 + k));
        uint2 pk; pk.x = pack2(v.x, v.y); pk.y = pack2(v.z, v.w);
        *(uint2*)&BsT[n][k] = pk;
      }
    }
    __syncthreads();

    // ---- B fragments: two 16B pieces per lane -> two ds_load_b128
    //      lane<16: K=0..7 & 16..23 ; lane>=16: K=8..15 & 24..31 ; N = lane&15
    v16bf bF[2];
    #pragma unroll
    for (int nt = 0; nt < 2; nt++) {
      const int col = waveN * 32 + nt * 16 + lrow;
      const uint4* pb = reinterpret_cast<const uint4*>(&BsT[col][0]);
      Frag32 f; f.lo = pb[half]; f.hi = pb[2 + half];
      bF[nt] = __builtin_bit_cast(v16bf, f);
    }
    // ---- A fragments + WMMA: two ds_load_b128 per fragment
    //      lane<16: K=0..7 & 16..23 ; lane>=16: K=8..15 & 24..31 ; M = lane&15
    #pragma unroll
    for (int mt = 0; mt < 4; mt++) {
      const int row = waveM * 64 + mt * 16 + lrow;
      const uint4* pa = reinterpret_cast<const uint4*>(&As[row][0]);
      Frag32 f; f.lo = pa[half]; f.hi = pa[2 + half];
      v16bf aF = __builtin_bit_cast(v16bf, f);
      #pragma unroll
      for (int nt = 0; nt < 2; nt++) {
        acc[mt][nt] = __builtin_amdgcn_wmma_f32_16x16x32_bf16(
            false, aF, false, bF[nt], (short)0, acc[mt][nt], false, false);
      }
    }
    __syncthreads();
  }

  // ---- epilogue + store (C layout: VGPR r -> M = r + (lane>>4)*8, N = lane&15)
  float* cptr = p.c + (long)batch * p.cbs;
  const float* auxp = p.aux ? (p.aux + (long)batch * p.auxbs) : nullptr;
  #pragma unroll
  for (int mt = 0; mt < 4; mt++) {
    #pragma unroll
    for (int nt = 0; nt < 2; nt++) {
      const long col = nBase + waveN * 32 + nt * 16 + lrow;
      const long rw0 = mBase + waveM * 64 + mt * 16 + half * 8;
      #pragma unroll
      for (int r = 0; r < 8; r++) {
        const long row = rw0 + r;
        float v = acc[mt][nt][r];
        if (p.mode == 3) {
          v = __expf(v) * auxp[row * p.ldaux + col];
        } else {
          if (p.bias) v += p.bias[col];
          if (p.mode == 1) v = fmaxf(v, 0.0f);
          else if (p.mode == 2 && col >= p.relu_col) v = fmaxf(v, 0.0f);
        }
        cptr[row * p.ldc + col] = v;
      }
    }
  }
}

// ---- na row l2-norm over 128 features: one 128-thread block per row ----
__global__ __launch_bounds__(128)
void l2norm_rows_128(float* x) {
  __shared__ float red[128];
  const long row = blockIdx.x;
  const int  t   = threadIdx.x;
  float v = x[row * 128 + t];
  red[t] = v * v;
  __syncthreads();
  for (int s = 64; s > 0; s >>= 1) { if (t < s) red[t] += red[t + s]; __syncthreads(); }
  const float inv = 1.0f / fmaxf(sqrtf(red[0]), 1e-12f);
  x[row * 128 + t] = v * inv;
}

// ---- dp row l2-norm (axis=2): one 256-thread block per (b,i) row ----
__global__ __launch_bounds__(256)
void dp_rownorm(float* dp, int N) {
  __shared__ float red[256];
  float* p = dp + (long)blockIdx.x * N;
  const int t = threadIdx.x;
  float s = 0.0f;
  for (int j = t; j < N; j += 256) { float v = p[j]; s += v * v; }
  red[t] = s;
  __syncthreads();
  for (int k = 128; k > 0; k >>= 1) { if (t < k) red[t] += red[t + k]; __syncthreads(); }
  const float inv = 1.0f / fmaxf(sqrtf(red[0]), 1e-12f);
  for (int j = t; j < N; j += 256) p[j] *= inv;
}

// ---- dp col l2-norm (axis=1): thread-per-column, coalesced row sweeps ----
__global__ __launch_bounds__(256)
void dp_colnorm(float* dp, int N) {
  const int j = blockIdx.x * 256 + threadIdx.x;
  float* p = dp + (long)blockIdx.y * N * N;
  float s = 0.0f;
  for (int i = 0; i < N; i++) { float v = p[(long)i * N + j]; s += v * v; }
  const float inv = 1.0f / fmaxf(sqrtf(s), 1e-12f);
  for (int i = 0; i < N; i++) p[(long)i * N + j] *= inv;
}

// ---------------------------------------------------------------------------
// Host orchestration
// ---------------------------------------------------------------------------
static inline void setSeg(GemmParams& g, int i, const float* p, long ld, long bs,
                          int klen, int relu) {
  g.a[i] = p; g.lda[i] = ld; g.abs_[i] = bs; g.klen[i] = klen; g.arelu[i] = relu;
}
static inline void launch_gemm(const GemmParams& g, int batch, hipStream_t s) {
  dim3 grid(g.N / BN, g.M / BM, batch);
  hipLaunchKernelGGL(wmma_gemm, grid, dim3(GTHREADS), 0, s, g);
}

extern "C" void kernel_launch(void* const* d_in, const int* in_sizes, int n_in,
                              void* d_out, int out_size, void* d_ws, size_t ws_size,
                              hipStream_t stream) {
  (void)in_sizes; (void)n_in; (void)out_size; (void)ws_size;
  const int B = 2, Nn = 2048, R = 128, TR = 384;

  const float* x   = (const float*)d_in[0];
  const float* adj = (const float*)d_in[1];
  auto P = [&](int L, int idx) { return (const float*)d_in[2 + L * 10 + idx]; };
  // idx: 0 rep_w 1 rep_b 2 addr_w 3 addr_b 4 nid_w 5 nid_b 6 att_w 7 att_b 8 upd_w 9 upd_b

  // workspace layout
  float* ws = (float*)d_ws;
  size_t off = 0;
  auto alloc = [&](size_t n) { float* p = ws + off; off += n; return p; };
  const size_t NR  = (size_t)B * Nn * R;
  const size_t NTR = (size_t)B * Nn * TR;
  float* zbuf = alloc(NR);
  float* rep  = alloc(NR);
  float* na   = alloc(NR);
  float* nid  = alloc(NR);
  float* src  = alloc(NR);
  float* lo0  = alloc(NR);
  float* lo1  = alloc(NR);
  float* lo2  = alloc(NR);
  float* cat0 = alloc(NTR);
  float* cat1 = alloc(NTR);
  float* cat2 = alloc(NTR);
  float* dp   = alloc((size_t)B * Nn * Nn);

  hipMemsetAsync(zbuf, 0, NR * sizeof(float), stream);

  // pyramid concat as multi-segment A (no copies)
  const float* parts[4][3] = { { x,   nullptr, nullptr },
                               { lo0, nullptr, nullptr },
                               { lo1, lo0,     nullptr },
                               { lo2, lo0,     x       } };
  const int nparts[4] = { 1, 1, 2, 3 };
  float* louts[4] = { lo0, lo1, lo2, (float*)d_out };

  for (int L = 0; L < 4; L++) {
    const int np = nparts[L];
    const int din = np * 128;

    // rep = X @ rep_w + b ; na_pre = X @ addr_w + b ; nid = relu(X @ nid_w + b)
    for (int which = 0; which < 3; which++) {
      GemmParams g{};
      for (int s = 0; s < np; s++) setSeg(g, s, parts[L][s], 128, 0, 128, 0);
      g.nseg = np;
      g.b    = P(L, which * 2);     g.ldb = R;
      g.bias = P(L, which * 2 + 1);
      g.c    = (which == 0) ? rep : (which == 1) ? na : nid;
      g.ldc  = R;
      g.M = B * Nn; g.N = R; g.K = din;
      g.mode = (which == 2) ? 1 : 0;
      launch_gemm(g, 1, stream);
    }
    hipLaunchKernelGGL(l2norm_rows_128, dim3(B * Nn), dim3(128), 0, stream, na);

    // dir 0: incoming loop (row-norm, dp @ rep)  -> final cat0
    // dir 1: outgoing loop (col-norm, dpT @ rep) -> final cat2
    for (int dir = 0; dir < 2; dir++) {
      float* catW[2] = { dir == 0 ? cat0 : cat2, cat1 };
      for (int t = 0; t < 3; t++) {
        float* cur = catW[t & 1];
        const float* pAddr; const float* pHid; long pLd;
        if (t == 0) { pAddr = zbuf; pHid = zbuf; pLd = R; }
        else { float* pc = catW[(t - 1) & 1]; pAddr = pc + R; pHid = pc + 2 * R; pLd = TR; }
        const long pBs = (t == 0) ? (long)Nn * R : (long)Nn * TR;

        // dp = exp(addr @ na^T) * adj   (batched, fused mask epilogue)
        GemmParams sg{};
        setSeg(sg, 0, pAddr, pLd, pBs, R, 0); sg.nseg = 1;
        sg.b = na; sg.ldb = R; sg.bbs = (long)Nn * R; sg.transB = 1;
        sg.c = dp; sg.ldc = Nn; sg.cbs = (long)Nn * Nn;
        sg.aux = adj; sg.ldaux = Nn; sg.auxbs = (long)Nn * Nn;
        sg.M = Nn; sg.N = Nn; sg.K = R; sg.mode = 3;
        launch_gemm(sg, B, stream);

        if (dir == 0)
          hipLaunchKernelGGL(dp_rownorm, dim3(B * Nn), dim3(256), 0, stream, dp, Nn);
        else
          hipLaunchKernelGGL(dp_colnorm, dim3(Nn / 256, B), dim3(256), 0, stream, dp, Nn);

        // src = dp @ rep (dir 0)  |  src = dp^T @ rep (dir 1)   (batched)
        GemmParams ag{};
        setSeg(ag, 0, dp, Nn, (long)Nn * Nn, Nn, 0); ag.nseg = 1;
        ag.transA = (dir == 1);
        ag.b = rep; ag.ldb = R; ag.bbs = (long)Nn * R;
        ag.c = src; ag.ldc = R; ag.cbs = (long)Nn * R;
        ag.M = Nn; ag.N = R; ag.K = Nn; ag.mode = 0;
        launch_gemm(ag, B, stream);

        // cat_out = [src, addr, hidden] @ att_w + att_b ; relu on last third
        GemmParams tg{};
        setSeg(tg, 0, src,   R,   0, R, 0);
        setSeg(tg, 1, pAddr, pLd, 0, R, 0);
        setSeg(tg, 2, pHid,  pLd, 0, R, 0);
        tg.nseg = 3;
        tg.b = P(L, 6); tg.ldb = TR; tg.bias = P(L, 7);
        tg.c = cur; tg.ldc = TR;
        tg.M = B * Nn; tg.N = TR; tg.K = TR;
        tg.mode = 2; tg.relu_col = 2 * R;
        launch_gemm(tg, 1, stream);
      }
    }

    // out = relu([relu(in_agg), nid, relu(out_agg)] @ upd_w + upd_b)
    GemmParams ug{};
    setSeg(ug, 0, cat0, TR, 0, R, 1);   // relu(in_agg) on load
    setSeg(ug, 1, nid,  R,  0, R, 0);   // nid already relu'd
    setSeg(ug, 2, cat2, TR, 0, R, 1);   // relu(out_agg) on load
    ug.nseg = 3;
    ug.b = P(L, 8); ug.ldb = 128; ug.bias = P(L, 9);
    ug.c = louts[L]; ug.ldc = 128;
    ug.M = B * Nn; ug.N = 128; ug.K = TR; ug.mode = 1;
    launch_gemm(ug, 1, stream);
  }
}
